// CFConv_75479755260176
// MI455X (gfx1250) — matrix operand order repaired and checked by reading, not compiled, not run on hardware
//
#include <hip/hip_runtime.h>
#include <hip/hip_bf16.h>

#define HID   128
#define NRBF  64
#define WAVES 8

typedef __attribute__((ext_vector_type(16))) __bf16 v16bf;
typedef __attribute__((ext_vector_type(8)))  __bf16 v8bf;
typedef __attribute__((ext_vector_type(8)))  float  v8f;

// ---------------------------------------------------------------------------
// WMMA wrappers: D = A(16x32 bf16) * B(32x16 bf16) + C(16x16 f32)
// The "reuse" variant sets the RA (reuse-A) operand hint: legal because it is
// only issued when the previous matrix instruction is the identical opcode
// with the same A operand (ISA 7.12, OPSEL[0]=RA). Literal immediates only.
// ---------------------------------------------------------------------------
__device__ __forceinline__ v8f wmma_bf16(v16bf a, v16bf b, v8f c) {
    return __builtin_amdgcn_wmma_f32_16x16x32_bf16(
        false, a, false, b, (short)0, c, /*reuse_a=*/false, /*reuse_b=*/false);
}
__device__ __forceinline__ v8f wmma_bf16_ra(v16bf a, v16bf b, v8f c) {
    return __builtin_amdgcn_wmma_f32_16x16x32_bf16(
        false, a, false, b, (short)0, c, /*reuse_a=*/true, /*reuse_b=*/false);
}

// A-fragment loader from a row-major f32 row (converted to bf16).
// 16-bit A 16x32 layout: lanes 0-15 hold K in {0..7, 16..23} (khalf=0),
// lanes 16-31 hold K in {8..15, 24..31} (khalf=8).
__device__ __forceinline__ v16bf load_a_f32(const float* __restrict__ row,
                                            int kbase, int khalf) {
    const float4 u0 = *(const float4*)(row + kbase + khalf);
    const float4 u1 = *(const float4*)(row + kbase + khalf + 4);
    const float4 u2 = *(const float4*)(row + kbase + 16 + khalf);
    const float4 u3 = *(const float4*)(row + kbase + 16 + khalf + 4);
    v16bf a;
    a[0]  = (__bf16)u0.x; a[1]  = (__bf16)u0.y; a[2]  = (__bf16)u0.z; a[3]  = (__bf16)u0.w;
    a[4]  = (__bf16)u1.x; a[5]  = (__bf16)u1.y; a[6]  = (__bf16)u1.z; a[7]  = (__bf16)u1.w;
    a[8]  = (__bf16)u2.x; a[9]  = (__bf16)u2.y; a[10] = (__bf16)u2.z; a[11] = (__bf16)u2.w;
    a[12] = (__bf16)u3.x; a[13] = (__bf16)u3.y; a[14] = (__bf16)u3.z; a[15] = (__bf16)u3.w;
    return a;
}

// A-fragment loader from a bf16 row staged in LDS (two 16-byte ds loads).
__device__ __forceinline__ v16bf load_a_lds(const __bf16* row, int kbase, int khalf) {
    v8bf lo = *(const v8bf*)(row + kbase + khalf);
    v8bf hi = *(const v8bf*)(row + kbase + 16 + khalf);
    v16bf a;
#pragma unroll
    for (int j = 0; j < 8; ++j) { a[j] = lo[j]; a[j + 8] = hi[j]; }
    return a;
}

// B fragments pre-packed fragment-linear: tile (ktile,ntile) = 512 bf16,
// lane ln owns 16 contiguous bf16 (32 bytes) -> one clean pair of b128 loads.
__device__ __forceinline__ v16bf load_b(const __bf16* __restrict__ packed,
                                        int ktile, int ntile, int ln) {
    return *(const v16bf*)(packed + (size_t)(ktile * 8 + ntile) * 512 + ln * 16);
}

__device__ __forceinline__ float silu_f(float x) {
    return x / (1.0f + __expf(-x));
}

// One k-step of a 16x128 output row-panel: 8 n-tiles against one A fragment.
// First WMMA fetches A; the remaining 7 use the operand-reuse hint.
__device__ __forceinline__ void gemm_kstep(v8f (&acc)[8], v16bf a,
                                           const __bf16* __restrict__ pW,
                                           int kt, int ln) {
    acc[0] = wmma_bf16(a, load_b(pW, kt, 0, ln), acc[0]);
#pragma unroll
    for (int nt = 1; nt < 8; ++nt)
        acc[nt] = wmma_bf16_ra(a, load_b(pW, kt, nt, ln), acc[nt]);
}

// ---------------------------------------------------------------------------
// Pack a [K x 128] f32 weight into bf16 WMMA-B fragment-linear order.
// B(32x16) layout: N = lane&15, lanes 0-15 hold K=kt*32+{0..7,16..23},
// lanes 16-31 hold K=kt*32+{8..15,24..31}.
// ---------------------------------------------------------------------------
__global__ void pack_weight_kernel(const float* __restrict__ W,
                                   __bf16* __restrict__ dst, int ktiles) {
    int tid = blockIdx.x * blockDim.x + threadIdx.x;
    int total = ktiles * 8 * 512;
    if (tid >= total) return;
    int j  = tid & 15;
    int ln = (tid >> 4) & 31;
    int t  = tid >> 9;            // ktile*8 + ntile
    int ntile = t & 7, ktile = t >> 3;
    int n = ntile * 16 + (ln & 15);
    int k = ktile * 32 + ((ln >> 4) * 8) + (j & 7) + ((j >> 3) * 16);
    dst[tid] = (__bf16)W[(size_t)k * HID + n];
}

// ---------------------------------------------------------------------------
// Edge kernel: one 16-edge tile per wave.
//   t1 = silu(rbf@We1 + be1)        (2 k-steps x 8 n-tiles WMMA)
//   w  = t1@We2 (+be2 folded later) (4 x 8 WMMA, t1 re-layouted via LDS)
//   g  = h[src]@Wlin                (4 x 8 WMMA, gathered rows, prefetched)
//   atomicAdd(agg[dst], w*g)
// ---------------------------------------------------------------------------
__global__ __launch_bounds__(32 * WAVES) void cfconv_edge_kernel(
    const float* __restrict__ rbf, const float* __restrict__ h,
    const int* __restrict__ ei,                  // [2, E]
    const __bf16* __restrict__ pWe1, const __bf16* __restrict__ pWe2,
    const __bf16* __restrict__ pWlin,
    const float* __restrict__ be1, const float* __restrict__ be2,
    float* __restrict__ agg, int E) {
    __shared__ __attribute__((aligned(32))) __bf16 lds[WAVES][16 * HID];

    const int ln    = threadIdx.x & 31;
    const int wv    = threadIdx.x >> 5;
    const int tile  = blockIdx.x * WAVES + wv;
    const int nTile = E >> 4;
    if (tile >= nTile) return;                   // wave-uniform exit, no barriers used

    const int e0    = tile << 4;
    const int mrow  = ln & 15;                   // A-fragment row this lane feeds
    const int khalf = (ln >> 4) << 3;            // 0 / 8 : K-half select for A
    const int ncol  = ln & 15;                   // C/D column this lane holds
    const int mbase = (ln >> 4) << 3;            // C/D rows: mbase..mbase+7
    __bf16* myLds   = &lds[wv][0];
    const v8f vzero = {};

    // Gather index for this lane's row; prefetch the h row now so the random
    // 512B gather hides behind the edge-filter GEMMs (global_prefetch_b8).
    const int src = ei[e0 + mrow];
    const float* __restrict__ hRow = h + (size_t)src * HID;
    __builtin_prefetch(hRow,      0, 0);
    __builtin_prefetch(hRow + 32, 0, 0);
    __builtin_prefetch(hRow + 64, 0, 0);
    __builtin_prefetch(hRow + 96, 0, 0);

    // ---- GEMM 1: t1 = silu(rbf_tile @ We1 + be1), K = 64 ----
    v8f acc[8];
#pragma unroll
    for (int nt = 0; nt < 8; ++nt) acc[nt] = vzero;
    const float* __restrict__ rbfRow = rbf + (size_t)(e0 + mrow) * NRBF;
#pragma unroll
    for (int kt = 0; kt < 2; ++kt)
        gemm_kstep(acc, load_a_f32(rbfRow, kt * 32, khalf), pWe1, kt, ln);

    // bias + SiLU, stage as bf16 [16 x 128] row-major tile in LDS (C->A relayout)
#pragma unroll
    for (int nt = 0; nt < 8; ++nt) {
        float bia = be1[nt * 16 + ncol];
#pragma unroll
        for (int i = 0; i < 8; ++i) {
            float x = acc[nt][i] + bia;
            myLds[(mbase + i) * HID + nt * 16 + ncol] = (__bf16)silu_f(x);
        }
    }
    asm volatile("s_wait_dscnt 0x0" ::: "memory");   // DS writes visible to same-wave DS reads

    // ---- GEMM 2: w = t1 @ We2, K = 128 ----
    v8f wacc[8];
#pragma unroll
    for (int nt = 0; nt < 8; ++nt) wacc[nt] = vzero;
    const __bf16* t1Row = myLds + mrow * HID;
#pragma unroll
    for (int kt = 0; kt < 4; ++kt)
        gemm_kstep(wacc, load_a_lds(t1Row, kt * 32, khalf), pWe2, kt, ln);

    // ---- GEMM 3: g = h[src] @ Wlin, K = 128 (gathered rows) ----
    v8f gacc[8];
#pragma unroll
    for (int nt = 0; nt < 8; ++nt) gacc[nt] = vzero;
#pragma unroll
    for (int kt = 0; kt < 4; ++kt)
        gemm_kstep(gacc, load_a_f32(hRow, kt * 32, khalf), pWlin, kt, ln);

    // ---- modulate + scatter-add (f32 atomics; agg working set lives in L2) ----
    int dsts[8];
#pragma unroll
    for (int i = 0; i < 8; ++i) dsts[i] = ei[E + e0 + mbase + i];
#pragma unroll
    for (int nt = 0; nt < 8; ++nt) {
        float bia = be2[nt * 16 + ncol];
#pragma unroll
        for (int i = 0; i < 8; ++i) {
            float val = (wacc[nt][i] + bia) * gacc[nt][i];
            atomicAdd(&agg[(size_t)dsts[i] * HID + nt * 16 + ncol], val);
        }
    }
}

// ---------------------------------------------------------------------------
// Node kernel: out = h + silu(agg@Wn1 + bn1)@Wn2 + bn2, 16 nodes per wave.
// ---------------------------------------------------------------------------
__global__ __launch_bounds__(32 * WAVES) void cfconv_node_kernel(
    const float* __restrict__ agg, const float* __restrict__ h,
    const __bf16* __restrict__ pWn1, const __bf16* __restrict__ pWn2,
    const float* __restrict__ bn1, const float* __restrict__ bn2,
    float* __restrict__ out, int N) {
    __shared__ __attribute__((aligned(32))) __bf16 lds[WAVES][16 * HID];

    const int ln    = threadIdx.x & 31;
    const int wv    = threadIdx.x >> 5;
    const int tile  = blockIdx.x * WAVES + wv;
    const int nTile = N >> 4;
    if (tile >= nTile) return;

    const int n0    = tile << 4;
    const int mrow  = ln & 15;
    const int khalf = (ln >> 4) << 3;
    const int ncol  = ln & 15;
    const int mbase = (ln >> 4) << 3;
    __bf16* myLds   = &lds[wv][0];
    const v8f vzero = {};

    // ---- GEMM 1: t = silu(agg_tile @ Wn1 + bn1) ----
    v8f acc[8];
#pragma unroll
    for (int nt = 0; nt < 8; ++nt) acc[nt] = vzero;
    const float* __restrict__ aRow = agg + (size_t)(n0 + mrow) * HID;
#pragma unroll
    for (int kt = 0; kt < 4; ++kt)
        gemm_kstep(acc, load_a_f32(aRow, kt * 32, khalf), pWn1, kt, ln);

#pragma unroll
    for (int nt = 0; nt < 8; ++nt) {
        float bia = bn1[nt * 16 + ncol];
#pragma unroll
        for (int i = 0; i < 8; ++i) {
            float x = acc[nt][i] + bia;
            myLds[(mbase + i) * HID + nt * 16 + ncol] = (__bf16)silu_f(x);
        }
    }
    asm volatile("s_wait_dscnt 0x0" ::: "memory");

    // ---- GEMM 2 + bias + residual ----
    v8f o[8];
#pragma unroll
    for (int nt = 0; nt < 8; ++nt) o[nt] = vzero;
    const __bf16* tRow = myLds + mrow * HID;
#pragma unroll
    for (int kt = 0; kt < 4; ++kt)
        gemm_kstep(o, load_a_lds(tRow, kt * 32, khalf), pWn2, kt, ln);

#pragma unroll
    for (int nt = 0; nt < 8; ++nt) {
        float bia = bn2[nt * 16 + ncol];
#pragma unroll
        for (int i = 0; i < 8; ++i) {
            size_t idx = (size_t)(n0 + mbase + i) * HID + nt * 16 + ncol;
            out[idx] = h[idx] + o[nt][i] + bia;
        }
    }
}

// ---------------------------------------------------------------------------
extern "C" void kernel_launch(void* const* d_in, const int* in_sizes, int n_in,
                              void* d_out, int out_size, void* d_ws, size_t ws_size,
                              hipStream_t stream) {
    const float* h    = (const float*)d_in[0];
    const float* rbf  = (const float*)d_in[1];
    const int*   ei   = (const int*)  d_in[2];
    const float* We1  = (const float*)d_in[3];
    const float* be1  = (const float*)d_in[4];
    const float* We2  = (const float*)d_in[5];
    const float* be2  = (const float*)d_in[6];
    const float* Wlin = (const float*)d_in[7];
    const float* Wn1  = (const float*)d_in[8];
    const float* bn1  = (const float*)d_in[9];
    const float* Wn2  = (const float*)d_in[10];
    const float* bn2  = (const float*)d_in[11];
    float* out = (float*)d_out;

    const int N = in_sizes[0] / HID;   // 50000
    const int E = in_sizes[2] / 2;     // 600000

    // workspace: [agg f32 N*HID | packed bf16 weights (144 KB)]
    float* agg = (float*)d_ws;
    size_t aggBytes = ((size_t)N * HID * sizeof(float) + 255) & ~(size_t)255;
    __bf16* pWe1  = (__bf16*)((char*)d_ws + aggBytes);
    __bf16* pWe2  = pWe1  + 2 * 8 * 512;
    __bf16* pWlin = pWe2  + 4 * 8 * 512;
    __bf16* pWn1  = pWlin + 4 * 8 * 512;
    __bf16* pWn2  = pWn1  + 4 * 8 * 512;

    hipMemsetAsync(agg, 0, (size_t)N * HID * sizeof(float), stream);

    pack_weight_kernel<<<(2 * 8 * 512 + 255) / 256, 256, 0, stream>>>(We1,  pWe1,  2);
    pack_weight_kernel<<<(4 * 8 * 512 + 255) / 256, 256, 0, stream>>>(We2,  pWe2,  4);
    pack_weight_kernel<<<(4 * 8 * 512 + 255) / 256, 256, 0, stream>>>(Wlin, pWlin, 4);
    pack_weight_kernel<<<(4 * 8 * 512 + 255) / 256, 256, 0, stream>>>(Wn1,  pWn1,  4);
    pack_weight_kernel<<<(4 * 8 * 512 + 255) / 256, 256, 0, stream>>>(Wn2,  pWn2,  4);

    const int eTiles = E / 16;
    cfconv_edge_kernel<<<(eTiles + WAVES - 1) / WAVES, 32 * WAVES, 0, stream>>>(
        rbf, h, ei, pWe1, pWe2, pWlin, be1, be2, agg, E);

    const int nTiles = N / 16;
    cfconv_node_kernel<<<(nTiles + WAVES - 1) / WAVES, 32 * WAVES, 0, stream>>>(
        agg, h, pWn1, pWn2, bn1, bn2, out, N);
}
